// Layer_52218212385325
// MI455X (gfx1250) — matrix operand order repaired and checked by reading, not compiled, not run on hardware
//
#include <hip/hip_runtime.h>

// ---------------------------------------------------------------------------
// Types / WMMA helpers (CDNA5 gfx1250, wave32)
// ---------------------------------------------------------------------------
typedef __attribute__((ext_vector_type(16))) _Float16    v16h;
typedef __attribute__((ext_vector_type(8)))  float       v8f;
typedef __attribute__((ext_vector_type(4)))  unsigned int u32x4;

union Frag {
    unsigned int u[8];
    u32x4        q[2];
    v16h         v;
};

__device__ __forceinline__ v8f wmma16(v16h a, v16h b, v8f c) {
    // D = A(16x32 f16) * B(32x16 f16) + C(16x16 f32)
    return __builtin_amdgcn_wmma_f32_16x16x32_f16(false, a, false, b,
                                                  (short)0, c, false, false);
}

// A-matrix 16x32 f16 K-pattern: contiguous, lane half h holds K 16h..16h+15
__device__ __forceinline__ int apat(int r, int khalf) {
    return 2 * r + 16 * khalf;
}

// ---------------------------------------------------------------------------
// Weight pack: W[k][n] fp32  ->  transposed split-f16 Whi/Wlo [n][k]
// ---------------------------------------------------------------------------
__global__ void pack_weight(const float* __restrict__ W,
                            _Float16* __restrict__ Whi,
                            _Float16* __restrict__ Wlo,
                            int K, int N) {
    long total = (long)K * N;
    for (long i = (long)blockIdx.x * blockDim.x + threadIdx.x; i < total;
         i += (long)gridDim.x * blockDim.x) {
        int k = (int)(i / N);
        int n = (int)(i % N);
        float v = W[i];
        _Float16 hi = (_Float16)v;
        size_t o = (size_t)n * K + k;
        Whi[o] = hi;
        Wlo[o] = (_Float16)(v - (float)hi);
    }
}

// ---------------------------------------------------------------------------
// RMSNorm + split-f16 pack: one block per row (D = 1024)
// ---------------------------------------------------------------------------
__global__ __launch_bounds__(256) void rmsnorm_pack(const float* __restrict__ X,
                                                    const float* __restrict__ g,
                                                    _Float16* __restrict__ Hhi,
                                                    _Float16* __restrict__ Hlo) {
    const int D = 1024;
    int row = blockIdx.x;
    const float* xr = X + (size_t)row * D;
    float ss = 0.f;
    for (int i = threadIdx.x; i < D; i += 256) {
        float v = xr[i];
        ss += v * v;
    }
    #pragma unroll
    for (int off = 16; off >= 1; off >>= 1) ss += __shfl_xor(ss, off, 32);
    __shared__ float wsum[8];
    if ((threadIdx.x & 31) == 0) wsum[threadIdx.x >> 5] = ss;
    __syncthreads();
    float tot = 0.f;
    #pragma unroll
    for (int j = 0; j < 8; ++j) tot += wsum[j];
    float scale = rsqrtf(tot * (1.0f / 1024.0f) + 1.1920929e-07f);
    for (int i = threadIdx.x; i < D; i += 256) {
        float v = xr[i] * scale * g[i];
        _Float16 hi = (_Float16)v;
        size_t o = (size_t)row * D + i;
        Hhi[o] = hi;
        Hlo[o] = (_Float16)(v - (float)hi);
    }
}

// ---------------------------------------------------------------------------
// V transpose pack: qkv16 f16 [B*T,3072] -> Vtg f16 [B][H][hd][T]
// (makes PV B-fragments straight dword loads: consecutive keys contiguous)
// ---------------------------------------------------------------------------
__global__ void pack_vt(const _Float16* __restrict__ qkv16,
                        _Float16* __restrict__ Vtg) {
    const long total = (long)2 * 16 * 64 * 2048;
    for (long i = (long)blockIdx.x * blockDim.x + threadIdx.x; i < total;
         i += (long)gridDim.x * blockDim.x) {
        int t = (int)(i & 2047);
        long rest = i >> 11;
        int dim = (int)(rest & 63);
        rest >>= 6;
        int h = (int)(rest & 15);
        int b = (int)(rest >> 4);
        Vtg[i] = qkv16[(size_t)(b * 2048 + t) * 3072 + 2048 + h * 64 + dim];
    }
}

// ---------------------------------------------------------------------------
// Split-f16 WMMA GEMM, 2x2 register-blocked (32x32 output per wave).
//   EPI 0: fp32 out              EPI 1: fp32 out + residual
//   EPI 2: GELU -> split f16     EPI 3: plain f16 out (qkv for attention)
// acc = Ah*Bh + Ah*Bl + Al*Bh per tile (fp32-faithful split product).
// ---------------------------------------------------------------------------
template <int EPI>
__global__ __launch_bounds__(128) void gemm_split(
    const _Float16* __restrict__ Ahi, const _Float16* __restrict__ Alo,
    const _Float16* __restrict__ Bhi, const _Float16* __restrict__ Blo,
    const float* __restrict__ bias, const float* __restrict__ resid,
    float* __restrict__ outf, _Float16* __restrict__ outhi,
    _Float16* __restrict__ outlo, int M, int N, int K) {
    int lane  = threadIdx.x;
    int nl    = lane & 15;
    int khalf = lane >> 4;
    int mt    = (blockIdx.y * 4 + threadIdx.y) * 2;  // two m tiles
    int nt    = blockIdx.x * 2;                      // two n tiles

    const _Float16* arh0 = Ahi + (size_t)(mt * 16 + nl) * K;
    const _Float16* arl0 = Alo + (size_t)(mt * 16 + nl) * K;
    const _Float16* arh1 = arh0 + (size_t)16 * K;
    const _Float16* arl1 = arl0 + (size_t)16 * K;
    const _Float16* brh0 = Bhi + (size_t)(nt * 16 + nl) * K;
    const _Float16* brl0 = Blo + (size_t)(nt * 16 + nl) * K;
    const _Float16* brh1 = brh0 + (size_t)16 * K;
    const _Float16* brl1 = brl0 + (size_t)16 * K;

    v8f c00 = {}, c01 = {}, c10 = {}, c11 = {};
    for (int kb = 0; kb < K; kb += 32) {
        Frag ah0, al0, ah1, al1, bh0, bl0, bh1, bl1;
        int ao = kb + 8 * khalf;   // A pattern: 4 contiguous dwords per half
        int bo = kb + 8 * khalf;   // B pattern: 4 contiguous dwords per half
        ah0.q[0] = *(const u32x4*)(arh0 + ao);
        ah0.q[1] = *(const u32x4*)(arh0 + ao + 16);
        al0.q[0] = *(const u32x4*)(arl0 + ao);
        al0.q[1] = *(const u32x4*)(arl0 + ao + 16);
        ah1.q[0] = *(const u32x4*)(arh1 + ao);
        ah1.q[1] = *(const u32x4*)(arh1 + ao + 16);
        al1.q[0] = *(const u32x4*)(arl1 + ao);
        al1.q[1] = *(const u32x4*)(arl1 + ao + 16);
        bh0.q[0] = *(const u32x4*)(brh0 + bo);
        bh0.q[1] = *(const u32x4*)(brh0 + bo + 16);
        bl0.q[0] = *(const u32x4*)(brl0 + bo);
        bl0.q[1] = *(const u32x4*)(brl0 + bo + 16);
        bh1.q[0] = *(const u32x4*)(brh1 + bo);
        bh1.q[1] = *(const u32x4*)(brh1 + bo + 16);
        bl1.q[0] = *(const u32x4*)(brl1 + bo);
        bl1.q[1] = *(const u32x4*)(brl1 + bo + 16);

        c00 = wmma16(ah0.v, bh0.v, c00);
        c00 = wmma16(ah0.v, bl0.v, c00);
        c00 = wmma16(al0.v, bh0.v, c00);
        c01 = wmma16(ah0.v, bh1.v, c01);
        c01 = wmma16(ah0.v, bl1.v, c01);
        c01 = wmma16(al0.v, bh1.v, c01);
        c10 = wmma16(ah1.v, bh0.v, c10);
        c10 = wmma16(ah1.v, bl0.v, c10);
        c10 = wmma16(al1.v, bh0.v, c10);
        c11 = wmma16(ah1.v, bh1.v, c11);
        c11 = wmma16(ah1.v, bl1.v, c11);
        c11 = wmma16(al1.v, bh1.v, c11);
    }

    #pragma unroll
    for (int i = 0; i < 2; ++i) {
        #pragma unroll
        for (int j = 0; j < 2; ++j) {
            v8f cc = (i == 0) ? (j == 0 ? c00 : c01) : (j == 0 ? c10 : c11);
            #pragma unroll
            for (int r = 0; r < 8; ++r) {
                int m = (mt + i) * 16 + r + 8 * khalf;
                int n = (nt + j) * 16 + nl;
                float v = cc[r] + bias[n];
                size_t oi = (size_t)m * N + n;
                if (EPI == 0) {
                    outf[oi] = v;
                } else if (EPI == 1) {
                    outf[oi] = v + resid[oi];
                } else if (EPI == 2) {
                    float gg = 0.5f * v * (1.0f + erff(v * 0.7071067811865475f));
                    _Float16 hi = (_Float16)gg;
                    outhi[oi] = hi;
                    outlo[oi] = (_Float16)(gg - (float)hi);
                } else {
                    outhi[oi] = (_Float16)v;
                }
            }
        }
    }
}

// ---------------------------------------------------------------------------
// Flash-style causal attention, one wave per (b, h, 16-query tile).
// qkv16: f16 [B*T, 3072] (q | k | v per head of 64); Vtg: f16 [B][H][64][T].
// Output scaled by d^-0.5 (faithful quirk: scale on softmax OUTPUT).
// Writes split-f16 O for the following proj GEMM.
// ---------------------------------------------------------------------------
__global__ __launch_bounds__(32) void attn_kernel(
    const _Float16* __restrict__ qkv16, const _Float16* __restrict__ Vtg,
    _Float16* __restrict__ Ohi, _Float16* __restrict__ Olo) {
    const int T = 2048, QLD = 3072, D = 1024, HD = 64;
    int qt = blockIdx.x, hh = blockIdx.y, b = blockIdx.z;
    int lane = threadIdx.x;
    int nl = lane & 15, khalf = lane >> 4;
    int baser = b * T;

    __shared__ _Float16 Pl[16 * 32];  // [qrow][key-in-block]

    // Q A-fragments for hd=64 (two K=32 steps): straight dword loads
    Frag qf[2];
    {
        const _Float16* qrow =
            qkv16 + (size_t)(baser + qt * 16 + nl) * QLD + hh * HD;
        #pragma unroll
        for (int s = 0; s < 2; ++s)
            #pragma unroll
            for (int r = 0; r < 8; ++r)
                qf[s].u[r] =
                    *(const unsigned int*)(qrow + s * 32 + apat(r, khalf));
    }
    const _Float16* vbase = Vtg + (size_t)(b * 16 + hh) * 64 * 2048;

    v8f o0 = {}, o1 = {}, o2 = {}, o3 = {};
    float mrun[8], lrun[8];
    #pragma unroll
    for (int r = 0; r < 8; ++r) { mrun[r] = -1e30f; lrun[r] = 0.f; }

    int kend = qt * 16 + 16;  // causal: keys [0, q]
    for (int kb = 0; kb < kend; kb += 32) {
        // S = Q * K^T for two 16-key tiles (K loads: contiguous dims)
        v8f s01[2];
        #pragma unroll
        for (int t = 0; t < 2; ++t) {
            Frag kf0, kf1;
            const _Float16* krow =
                qkv16 + (size_t)(baser + kb + 16 * t + nl) * QLD + D + hh * HD;
            #pragma unroll
            for (int r = 0; r < 8; ++r) {
                int d0 = 16 * khalf + 2 * r;  // B 32x16 pattern
                kf0.u[r] = *(const unsigned int*)(krow + d0);
                kf1.u[r] = *(const unsigned int*)(krow + 32 + d0);
            }
            v8f sv = {};
            sv = wmma16(qf[0].v, kf0.v, sv);
            sv = wmma16(qf[1].v, kf1.v, sv);
            #pragma unroll
            for (int r = 0; r < 8; ++r) {  // causal mask
                int qrow_i = qt * 16 + r + 8 * khalf;
                int key = kb + 16 * t + nl;
                if (key > qrow_i) sv[r] = -1e30f;
            }
            s01[t] = sv;
        }

        // online softmax over the 32 keys (row lives in a 16-lane group)
        float scale[8];
        #pragma unroll
        for (int r = 0; r < 8; ++r) {
            float mloc = fmaxf(s01[0][r], s01[1][r]);
            #pragma unroll
            for (int off = 8; off >= 1; off >>= 1)
                mloc = fmaxf(mloc, __shfl_xor(mloc, off, 16));
            float mnew = fmaxf(mrun[r], mloc);
            scale[r] = __expf(mrun[r] - mnew);
            float p0 = __expf(s01[0][r] - mnew);
            float p1 = __expf(s01[1][r] - mnew);
            s01[0][r] = p0;
            s01[1][r] = p1;
            float rs = p0 + p1;
            #pragma unroll
            for (int off = 8; off >= 1; off >>= 1)
                rs += __shfl_xor(rs, off, 16);
            lrun[r] = lrun[r] * scale[r] + rs;
            mrun[r] = mnew;
        }
        #pragma unroll
        for (int r = 0; r < 8; ++r) {
            o0[r] *= scale[r]; o1[r] *= scale[r];
            o2[r] *= scale[r]; o3[r] *= scale[r];
        }

        // P (C-layout) -> LDS -> A-fragment layout
        __syncthreads();
        #pragma unroll
        for (int t = 0; t < 2; ++t)
            #pragma unroll
            for (int r = 0; r < 8; ++r) {
                int m = r + 8 * khalf;
                Pl[m * 32 + 16 * t + nl] = (_Float16)s01[t][r];
            }
        __syncthreads();
        Frag pf;
        #pragma unroll
        for (int r = 0; r < 8; ++r)
            pf.u[r] = *(const unsigned int*)&Pl[nl * 32 + apat(r, khalf)];

        // O += P * V : V B-fragments direct from transposed global copy
        #pragma unroll
        for (int dt = 0; dt < 4; ++dt) {
            Frag vf;
            const _Float16* vd = vbase + (size_t)(dt * 16 + nl) * 2048 + kb;
            #pragma unroll
            for (int r = 0; r < 8; ++r)
                vf.u[r] = *(const unsigned int*)(vd + 16 * khalf + 2 * r);
            v8f acc = (dt == 0) ? o0 : (dt == 1) ? o1 : (dt == 2) ? o2 : o3;
            acc = wmma16(pf.v, vf.v, acc);
            if (dt == 0) o0 = acc; else if (dt == 1) o1 = acc;
            else if (dt == 2) o2 = acc; else o3 = acc;
        }
    }

    // epilogue: O = softmax(S)*V / l * d^-0.5, write split-f16
    #pragma unroll
    for (int r = 0; r < 8; ++r) {
        int m = r + 8 * khalf;
        size_t row = (size_t)(baser + qt * 16 + m);
        float inv = 0.03125f / lrun[r];  // 1024^-0.5 = 1/32
        float vals[4] = { o0[r], o1[r], o2[r], o3[r] };
        #pragma unroll
        for (int dt = 0; dt < 4; ++dt) {
            float val = vals[dt] * inv;
            int col = hh * HD + dt * 16 + nl;
            _Float16 hi = (_Float16)val;
            Ohi[row * D + col] = hi;
            Olo[row * D + col] = (_Float16)(val - (float)hi);
        }
    }
}

// ---------------------------------------------------------------------------
// Host orchestration
// ---------------------------------------------------------------------------
extern "C" void kernel_launch(void* const* d_in, const int* in_sizes, int n_in,
                              void* d_out, int out_size, void* d_ws,
                              size_t ws_size, hipStream_t stream) {
    (void)in_sizes; (void)n_in; (void)out_size; (void)ws_size;
    const float* x      = (const float*)d_in[0];
    const float* w_attn = (const float*)d_in[1];
    const float* b_attn = (const float*)d_in[2];
    const float* w_proj = (const float*)d_in[3];
    const float* b_proj = (const float*)d_in[4];
    const float* w_fc1  = (const float*)d_in[5];
    const float* b_fc1  = (const float*)d_in[6];
    const float* w_fc2  = (const float*)d_in[7];
    const float* b_fc2  = (const float*)d_in[8];
    const float* g1     = (const float*)d_in[9];
    const float* g2     = (const float*)d_in[10];

    const int M = 4096;  // B*T
    const size_t sH    = (size_t)4096 * 1024 * 2;  // split-f16 activation buf
    const size_t sQKV  = (size_t)4096 * 3072 * 2;  // f16 qkv
    const size_t sG    = (size_t)4096 * 4096 * 2;
    const size_t sX    = (size_t)4096 * 1024 * 4;
    const size_t sVT   = (size_t)2 * 16 * 64 * 2048 * 2;

    char* ws = (char*)d_ws;
    size_t off = 0;
    auto alloc = [&](size_t sz) -> char* {
        char* p = ws + off;
        off += (sz + 255) & ~(size_t)255;
        return p;
    };
    _Float16* wah  = (_Float16*)alloc((size_t)1024 * 3072 * 2);
    _Float16* wal  = (_Float16*)alloc((size_t)1024 * 3072 * 2);
    _Float16* wph  = (_Float16*)alloc((size_t)1024 * 1024 * 2);
    _Float16* wpl  = (_Float16*)alloc((size_t)1024 * 1024 * 2);
    _Float16* wf1h = (_Float16*)alloc((size_t)1024 * 4096 * 2);
    _Float16* wf1l = (_Float16*)alloc((size_t)1024 * 4096 * 2);
    _Float16* wf2h = (_Float16*)alloc((size_t)4096 * 1024 * 2);
    _Float16* wf2l = (_Float16*)alloc((size_t)4096 * 1024 * 2);
    size_t regA_sz = 2 * sG;  // covers (h1 hi/lo + qkv16) and (g hi/lo)
    char* regionA = alloc(regA_sz);
    _Float16* h1hi  = (_Float16*)regionA;
    _Float16* h1lo  = (_Float16*)(regionA + sH);
    _Float16* qkv16 = (_Float16*)(regionA + 2 * sH);
    _Float16* ghi   = (_Float16*)regionA;          // reuse (h1,qkv dead)
    _Float16* glo   = (_Float16*)(regionA + sG);
    char* regionB = alloc(2 * sH);
    _Float16* ohi  = (_Float16*)regionB;
    _Float16* olo  = (_Float16*)(regionB + sH);
    _Float16* h2hi = ohi;                          // reuse (o dead)
    _Float16* h2lo = olo;
    float*    x1   = (float*)alloc(sX);
    _Float16* vtg  = (_Float16*)alloc(sVT);
    float*    out  = (float*)d_out;

    // 1) pack weights (transpose + split-f16)
    pack_weight<<<4096, 256, 0, stream>>>(w_attn, wah, wal, 1024, 3072);
    pack_weight<<<4096, 256, 0, stream>>>(w_proj, wph, wpl, 1024, 1024);
    pack_weight<<<4096, 256, 0, stream>>>(w_fc1, wf1h, wf1l, 1024, 4096);
    pack_weight<<<4096, 256, 0, stream>>>(w_fc2, wf2h, wf2l, 4096, 1024);

    dim3 blk(32, 4);
    // 2) h1 = rmsnorm(x, g1)
    rmsnorm_pack<<<M, 256, 0, stream>>>(x, g1, h1hi, h1lo);
    // 3) qkv16 = f16(h1 @ w_attn + b_attn)
    gemm_split<3><<<dim3(3072 / 32, M / 128), blk, 0, stream>>>(
        h1hi, h1lo, wah, wal, b_attn, nullptr, nullptr, qkv16, nullptr,
        M, 3072, 1024);
    // 3b) V transposed copy for PV fragments
    pack_vt<<<2048, 256, 0, stream>>>(qkv16, vtg);
    // 4) o = attention(qkv16)
    attn_kernel<<<dim3(2048 / 16, 16, 2), 32, 0, stream>>>(qkv16, vtg, ohi, olo);
    // 5) x1 = x + o @ w_proj + b_proj
    gemm_split<1><<<dim3(1024 / 32, M / 128), blk, 0, stream>>>(
        ohi, olo, wph, wpl, b_proj, x, x1, nullptr, nullptr, M, 1024, 1024);
    // 6) h2 = rmsnorm(x1, g2)
    rmsnorm_pack<<<M, 256, 0, stream>>>(x1, g2, h2hi, h2lo);
    // 7) g = gelu(h2 @ w_fc1 + b_fc1)  (split-f16 out)
    gemm_split<2><<<dim3(4096 / 32, M / 128), blk, 0, stream>>>(
        h2hi, h2lo, wf1h, wf1l, b_fc1, nullptr, nullptr, ghi, glo,
        M, 4096, 1024);
    // 8) out = x1 + g @ w_fc2 + b_fc2
    gemm_split<1><<<dim3(1024 / 32, M / 128), blk, 0, stream>>>(
        ghi, glo, wf2h, wf2l, b_fc2, x1, out, nullptr, nullptr, M, 1024, 4096);
}
